// NGP_63350767616160
// MI455X (gfx1250) — compile-verified, matrix operand order checked by reading, not run on hardware
//
#include <hip/hip_runtime.h>

typedef _Float16 v16h __attribute__((ext_vector_type(16)));
typedef _Float16 v8h  __attribute__((ext_vector_type(8)));
typedef float    v8f  __attribute__((ext_vector_type(8)));

#define N_HASH    524288u
#define HASH_MASK (N_HASH - 1u)
#define PRIME_C   2654435761u
#define H_DIM     32
#define M_PTS     1048576
#define WAVES_PER_BLOCK 8
#define TILES_PER_WAVE  4

// round(2^linspace(log2(16), log2(2048), 16)) — matches reference RESOLUTIONS
__device__ __constant__ float c_res[16] = {
    16.f, 22.f, 31.f, 42.f, 58.f, 81.f, 111.f, 154.f,
    213.f, 294.f, 406.f, 562.f, 776.f, 1072.f, 1482.f, 2048.f
};

__device__ __forceinline__ float lrelu(float x) {
    return fmaxf(x, 0.f) + 0.01f * fminf(x, 0.f);
}

// Build a 32x16 B-matrix fragment (f32 weights -> f16) for one 16-col tile.
// CDNA5 16-bit B layout: lane l holds col n=l%16; VGPR v / elem e holds
// K = ((v<4) ? (l<16?0:8) : (l<16?16:24)) + (v%4)*2 + e  -> K-contiguous groups.
__device__ __forceinline__ v16h load_b_frag(const float* __restrict__ W,
                                            int lane, int tile) {
    int n   = tile * 16 + (lane & 15);
    int kb0 = (lane < 16) ? 0  : 8;
    int kb1 = (lane < 16) ? 16 : 24;
    v16h b;
#pragma unroll
    for (int i = 0; i < 8; ++i) {
        b[i]     = (_Float16)W[(kb0 + i) * H_DIM + n];
        b[i + 8] = (_Float16)W[(kb1 + i) * H_DIM + n];
    }
    return b;
}

__global__ __launch_bounds__(256)
void ngp_fused_kernel(const float* __restrict__ coords,
                      const float* __restrict__ hashf,
                      const float* __restrict__ W0, const float* __restrict__ b0,
                      const float* __restrict__ W1, const float* __restrict__ b1,
                      const float* __restrict__ W2, const float* __restrict__ b2,
                      float* __restrict__ out) {
    __shared__ __align__(16) float smem[WAVES_PER_BLOCK * 512];

    const int tid  = threadIdx.x;
    const int wave = tid >> 5;
    const int lane = tid & 31;
    const int c    = lane & 15;          // row-in-tile (for A) / col (for D)
    const bool hi  = lane >= 16;

    float*    myf = smem + wave * 512;                 // 2KB f32 view (wave-private)
    _Float16* myh = reinterpret_cast<_Float16*>(myf);  // 1KB f16 view (same region)

    // Loop-invariant: weight fragments + per-column biases.
    const v16h B0a = load_b_frag(W0, lane, 0);
    const v16h B0b = load_b_frag(W0, lane, 1);
    const v16h B1a = load_b_frag(W1, lane, 0);
    const v16h B1b = load_b_frag(W1, lane, 1);
    const float bias00 = b0[c], bias01 = b0[16 + c];
    const float bias10 = b1[c], bias11 = b1[16 + c];

    const int rowoff = hi ? 8 : 0;   // D-frag row offset for this half-wave
    const int lvlA   = hi ? 4 : 0;   // A-frag K 0-7  <-> levels 0-3 / 4-7
    const int lvlB   = hi ? 12 : 8;  // A-frag K16-23 <-> levels 8-11 / 12-15

    // Per-lane resolutions for its 8 levels (j<4 -> lvlA+j, j>=4 -> lvlB+j-4).
    float res[8];
#pragma unroll
    for (int j = 0; j < 8; ++j)
        res[j] = c_res[(j < 4 ? lvlA + j : lvlB + j - 4)];

    const float2* tabs[8];
#pragma unroll
    for (int j = 0; j < 8; ++j)
        tabs[j] = reinterpret_cast<const float2*>(hashf) +
                  (size_t)(j < 4 ? lvlA + j : lvlB + j - 4) * N_HASH;

    const int tile0 = (blockIdx.x * WAVES_PER_BLOCK + wave) * TILES_PER_WAVE;

    for (int t = 0; t < TILES_PER_WAVE; ++t) {
        const int tile = tile0 + t;
        const int p    = tile * 16 + c;  // point index (lanes l, l+16 share it)
        const float2 xy = reinterpret_cast<const float2*>(coords)[p];

        // ---- Phase 1: all 8 levels' corner indices + lerp weights ----------
        float wxs[8], wys[8];
        unsigned i00[8], i01[8], i10[8], i11[8];
#pragma unroll
        for (int j = 0; j < 8; ++j) {
            float xs = xy.x * res[j], ys = xy.y * res[j];
            float xf = floorf(xs),   yf = floorf(ys);
            wxs[j] = xs - xf;  wys[j] = ys - yf;
            unsigned x0 = (unsigned)xf, y0 = (unsigned)yf;
            unsigned hy0 = PRIME_C * y0;
            unsigned hy1 = hy0 + PRIME_C;                 // PRIME_C*(y0+1) mod 2^32
            i00[j] = (x0        ^ hy0) & HASH_MASK;
            i01[j] = (x0        ^ hy1) & HASH_MASK;
            i10[j] = ((x0 + 1u) ^ hy0) & HASH_MASK;
            i11[j] = ((x0 + 1u) ^ hy1) & HASH_MASK;
        }

        // ---- Phase 2: issue all 32 independent L2 gathers (no consumers) ---
        float2 f00[8], f01[8], f10[8], f11[8];
#pragma unroll
        for (int j = 0; j < 8; ++j) {
            f00[j] = tabs[j][i00[j]];
            f01[j] = tabs[j][i01[j]];
            f10[j] = tabs[j][i10[j]];
            f11[j] = tabs[j][i11[j]];
        }

        // ---- Phase 3: bilerp reduction straight into the WMMA A-fragment --
        v16h a;
#pragma unroll
        for (int j = 0; j < 8; ++j) {
            float w00 = (1.f - wxs[j]) * (1.f - wys[j]);
            float w01 = (1.f - wxs[j]) * wys[j];
            float w10 = wxs[j] * (1.f - wys[j]);
            float w11 = wxs[j] * wys[j];
            float fx = f00[j].x * w00 + f01[j].x * w01 +
                       f10[j].x * w10 + f11[j].x * w11;
            float fy = f00[j].y * w00 + f01[j].y * w01 +
                       f10[j].y * w10 + f11[j].y * w11;
            a[2 * j]     = (_Float16)fx;
            a[2 * j + 1] = (_Float16)fy;
        }

        // --- Layer 0: (16x32 f16) x (32x32) via two WMMAs, f32 accumulate ---
        v8f acc_a = {}, acc_b = {};
        acc_a = __builtin_amdgcn_wmma_f32_16x16x32_f16(false, a, false, B0a,
                                                       (short)0, acc_a, false, false);
        acc_b = __builtin_amdgcn_wmma_f32_16x16x32_f16(false, a, false, B0b,
                                                       (short)0, acc_b, false, false);

        // bias + leaky-relu, stage f16 row-major [row][k] for next A-frag load
#pragma unroll
        for (int r = 0; r < 8; ++r) {
            myh[(r + rowoff) * 32 + c]      = (_Float16)lrelu(acc_a[r] + bias00);
            myh[(r + rowoff) * 32 + 16 + c] = (_Float16)lrelu(acc_b[r] + bias01);
        }

        // Reload as A-fragment: K-contiguous -> two aligned b128 LDS loads.
        // Wave-private region + per-wave in-order DS ops => no barrier needed.
        union { v16h v; v8h h[2]; } A1;
        A1.h[0] = *reinterpret_cast<const v8h*>(myh + c * 32 + (hi ? 8 : 0));
        A1.h[1] = *reinterpret_cast<const v8h*>(myh + c * 32 + (hi ? 24 : 16));

        // --- Layer 1 ---
        acc_a = (v8f){}; acc_b = (v8f){};
        acc_a = __builtin_amdgcn_wmma_f32_16x16x32_f16(false, A1.v, false, B1a,
                                                       (short)0, acc_a, false, false);
        acc_b = __builtin_amdgcn_wmma_f32_16x16x32_f16(false, A1.v, false, B1b,
                                                       (short)0, acc_b, false, false);

#pragma unroll
        for (int r = 0; r < 8; ++r) {
            myf[(r + rowoff) * 32 + c]      = lrelu(acc_a[r] + bias10);
            myf[(r + rowoff) * 32 + 16 + c] = lrelu(acc_b[r] + bias11);
        }

        // --- Layer 2 (32 -> 3): tiny, VALU on lanes 0-15, one row each ---
        if (!hi) {
            float o0 = b2[0], o1 = b2[1], o2 = b2[2];
            const float* hrow = myf + c * 32;
#pragma unroll
            for (int k = 0; k < 32; ++k) {
                const float hv = hrow[k];
                o0 = fmaf(hv, W2[k * 3 + 0], o0);
                o1 = fmaf(hv, W2[k * 3 + 1], o1);
                o2 = fmaf(hv, W2[k * 3 + 2], o2);
            }
            float* op = out + (size_t)p * 3;
            op[0] = o0; op[1] = o1; op[2] = o2;   // contiguous -> b96 store
        }
        // EXEC reconverges here; next iteration's WMMAs see all-ones again.
    }
}

extern "C" void kernel_launch(void* const* d_in, const int* in_sizes, int n_in,
                              void* d_out, int out_size, void* d_ws, size_t ws_size,
                              hipStream_t stream) {
    (void)in_sizes; (void)n_in; (void)d_ws; (void)ws_size; (void)out_size;
    const float* coords = (const float*)d_in[0];
    const float* hashf  = (const float*)d_in[1];
    const float* W0     = (const float*)d_in[2];
    const float* b0     = (const float*)d_in[3];
    const float* W1     = (const float*)d_in[4];
    const float* b1     = (const float*)d_in[5];
    const float* W2     = (const float*)d_in[6];
    const float* b2     = (const float*)d_in[7];
    float* out = (float*)d_out;

    const int tiles  = M_PTS / 16;                                   // 65536
    const int blocks = tiles / (WAVES_PER_BLOCK * TILES_PER_WAVE);   // 2048

    hipLaunchKernelGGL(ngp_fused_kernel, dim3(blocks), dim3(256), 0, stream,
                       coords, hashf, W0, b0, W1, b1, W2, b2, out);
}
